// AEConv9_7129645711755
// MI455X (gfx1250) — matrix operand order repaired and checked by reading, not compiled, not run on hardware
//
#include <hip/hip_runtime.h>

// ---------------------------------------------------------------------------
// CDNA5 (gfx1250) wave32 WMMA types / helpers
// ---------------------------------------------------------------------------
typedef __attribute__((ext_vector_type(16))) __bf16        bf16x16;
typedef __attribute__((ext_vector_type(8)))  float         f32x8;
typedef __attribute__((ext_vector_type(4)))  unsigned int  u32x4;

union FragAB {            // A or B operand of v_wmma_f32_16x16x32_bf16
  bf16x16      v;
  u32x4        q[2];
  unsigned int u[8];
};
union FragC {             // C/D accumulator
  f32x8 v;
  float f[8];
};

__device__ __forceinline__ unsigned short f2bf(float x) {
  unsigned int u = __builtin_bit_cast(unsigned int, x);
  u += 0x7fffu + ((u >> 16) & 1u);          // round-to-nearest-even
  return (unsigned short)(u >> 16);
}
__device__ __forceinline__ float bf2f(unsigned short h) {
  return __builtin_bit_cast(float, ((unsigned int)h) << 16);
}
__device__ __forceinline__ unsigned int pack2(unsigned short lo, unsigned short hi) {
  return (unsigned int)lo | ((unsigned int)hi << 16);
}
__device__ __forceinline__ void wmma_bf16(FragC& c, const FragAB& a, const FragAB& b) {
  // (neg_a, A, neg_b, B, c_mod, C, reuse_a, reuse_b)
  c.v = __builtin_amdgcn_wmma_f32_16x16x32_bf16(false, a.v, false, b.v,
                                                (short)0, c.v, false, false);
}

// Problem constants
#define BATCH 16
#define HW    128
#define KCH   256           // expert channels
#define DCH   128           // bottleneck dim
#define TAPS  81
#define TPAD  96            // taps padded to 3 K-steps of 32
#define NPIX  (BATCH * HW * HW)   // 262144

// ---------------------------------------------------------------------------
// Kernel 0: weight repack (bf16, [N][K] layout for B-fragments) + sigmoid +
//           zero usage counters.  All tiny; L2-resident afterwards.
// ---------------------------------------------------------------------------
__global__ void prep_kernel(const float* __restrict__ w_conv,   // [256,81]
                            const float* __restrict__ gate,     // [256]
                            const float* __restrict__ w_down,   // [128,256]
                            const float* __restrict__ w_up,     // [256,128]
                            unsigned short* __restrict__ wenc_b, // [256][96]  B: k=tap, n=ch
                            unsigned short* __restrict__ wup_b,  // [256][128] B: k=d,   n=ch
                            unsigned short* __restrict__ wf2_b,  // [96][256]  B: k=ch,  n=tap (flipped)
                            float* __restrict__ wdt,             // [256][128] w_down^T (f32)
                            float* __restrict__ sigg,            // [256]
                            unsigned int* __restrict__ ucnt)     // [256]
{
  int tid = blockIdx.x * blockDim.x + threadIdx.x;
  int nth = gridDim.x * blockDim.x;

  for (int i = tid; i < KCH; i += nth) {
    sigg[i] = 1.f / (1.f + __expf(-gate[i]));
    ucnt[i] = 0u;
  }
  for (int i = tid; i < KCH * TPAD; i += nth) {
    int ch = i / TPAD, t = i - ch * TPAD;
    wenc_b[ch * TPAD + t] = (t < TAPS) ? f2bf(w_conv[ch * TAPS + t]) : (unsigned short)0;
  }
  for (int i = tid; i < TPAD * KCH; i += nth) {
    int t = i / KCH, ch = i - t * KCH;
    unsigned short v = 0;
    if (t < TAPS) {
      int ty = t / 9, tx = t - ty * 9;
      v = f2bf(w_conv[ch * TAPS + (8 - ty) * 9 + (8 - tx)]);   // spatial flip
    }
    wf2_b[t * KCH + ch] = v;
  }
  for (int i = tid; i < KCH * DCH; i += nth) {
    int ch = i >> 7, d = i & 127;
    wup_b[i] = f2bf(w_up[i]);                 // w_up already [256][128] = [n][k]
    wdt[ch * DCH + d] = w_down[d * KCH + ch]; // transpose for row-wise access
  }
}

// ---------------------------------------------------------------------------
// Kernel 1: encoder. 32-pixel x-strip per workgroup (128 thr = 4 waves).
// im2col GEMM [32 x 96] x [96 x 256] via WMMA, gate+relu, per-pixel argmax.
// ---------------------------------------------------------------------------
#define ACS 264   // gated-activation LDS row stride (floats), padded

__global__ __launch_bounds__(128) void enc_kernel(
    const float* __restrict__ x,                  // [16,1,128,128]
    const unsigned short* __restrict__ wenc_b,    // [256][96] bf16
    const float* __restrict__ sigg,               // [256]
    int* __restrict__ idx_arr,                    // [NPIX]
    float* __restrict__ val_arr,                  // [NPIX]
    unsigned int* __restrict__ ucnt)              // [256]
{
  __shared__ __attribute__((aligned(16))) unsigned short Ai[32 * TPAD]; //  6 KB
  __shared__ __attribute__((aligned(16))) float          Aa[32 * ACS];  // 33 KB

  const int tid  = threadIdx.x;
  const int lane = tid & 31;
  const int wave = tid >> 5;
  const int half = lane >> 4;
  const int l15  = lane & 15;

  const int x0 = blockIdx.x * 32;   // 0..96
  const int y  = blockIdx.y;        // 0..127
  const int b  = blockIdx.z;        // 0..15
  const float* xb = x + b * (HW * HW);

  // im2col into LDS: Ai[m][t] = x[y+ty-4, x0+m+tx-4] (zero pad; taps 81..95 = 0)
  for (int i = tid; i < 32 * TPAD; i += 128) {
    int m = i / TPAD, t = i - m * TPAD;
    float v = 0.f;
    if (t < TAPS) {
      int ty = t / 9, txx = t - ty * 9;
      int gy = y + ty - 4;
      int gx = x0 + m + txx - 4;
      if (gy >= 0 && gy < HW && gx >= 0 && gx < HW) v = xb[gy * HW + gx];
    }
    Ai[m * TPAD + t] = f2bf(v);
  }
  __syncthreads();

  // GEMM: wave w handles channels [w*64, w*64+64), m-tiles {0,1}
  FragC acc[2][4];
#pragma unroll
  for (int mt = 0; mt < 2; mt++)
#pragma unroll
    for (int nt = 0; nt < 4; nt++)
#pragma unroll
      for (int r = 0; r < 8; r++) acc[mt][nt].f[r] = 0.f;

  for (int ks = 0; ks < 3; ks++) {
    int k0 = ks * 32;
    FragAB afr[2];
#pragma unroll
    for (int mt = 0; mt < 2; mt++) {
      const unsigned short* p = &Ai[(mt * 16 + l15) * TPAD + k0 + 8 * half];
      afr[mt].q[0] = *(const u32x4*)p;
      afr[mt].q[1] = *(const u32x4*)(p + 16);
    }
#pragma unroll
    for (int nt = 0; nt < 4; nt++) {
      int ch = wave * 64 + nt * 16 + l15;
      const unsigned short* p = wenc_b + ch * TPAD + k0 + 8 * half;
      FragAB bfr;
      bfr.q[0] = *(const u32x4*)p;
      bfr.q[1] = *(const u32x4*)(p + 16);
#pragma unroll
      for (int mt = 0; mt < 2; mt++) wmma_bf16(acc[mt][nt], afr[mt], bfr);
    }
  }

  // a = relu(z) * sigmoid(gate[ch]) -> LDS
#pragma unroll
  for (int mt = 0; mt < 2; mt++)
#pragma unroll
    for (int nt = 0; nt < 4; nt++) {
      int ch = wave * 64 + nt * 16 + l15;
      float g = sigg[ch];
#pragma unroll
      for (int r = 0; r < 8; r++) {
        int m = mt * 16 + r + 8 * half;
        float z = acc[mt][nt].f[r];
        Aa[m * ACS + ch] = (z > 0.f) ? z * g : 0.f;
      }
    }
  __syncthreads();

  // per-pixel argmax over 256 channels (first-max like jnp.argmax)
  if (tid < 32) {
    const float* row = &Aa[tid * ACS];
    float best = row[0];
    int bi = 0;
    for (int k = 1; k < KCH; k++) {
      float v = row[k];
      if (v > best) { best = v; bi = k; }
    }
    int gp = (b * HW + y) * HW + (x0 + tid);
    idx_arr[gp] = bi;
    val_arr[gp] = best;
    atomicAdd(&ucnt[bi], 1u);
  }
}

// ---------------------------------------------------------------------------
// Kernel 2: fused decoder.  Tile = 16(y) x 8(x) outputs, halo Q = 24x16 = 384.
//   h[q,d]   = relu(v_q * w_down^T[idx_q][d] + b_down[d])  (built in registers)
//   a2[q,ch] = relu(h[q,:] @ w_up^T + b_up[ch])            WMMA -> LDS bf16
//   s[q,t]   = a2[q,:] @ wf2[:,t]                          WMMA -> LDS bf16
//   xhat[p]  = sum_t s[p + off_t, t]                       gather (mask OOB)
// ---------------------------------------------------------------------------
#define QH   24
#define QW   16
#define QN   (QH * QW)     // 384 halo pixels = 24 m-tiles
#define A2S  264           // a2 LDS stride (bf16 elems)
#define SSS  104           // s  LDS stride (bf16 elems)

__global__ __launch_bounds__(256) void dec_kernel(
    const int* __restrict__ idx_arr,
    const float* __restrict__ val_arr,
    const float* __restrict__ wdt,               // [256][128] f32
    const float* __restrict__ b_down,            // [128]
    const unsigned short* __restrict__ wup_b,    // [256][128] bf16
    const float* __restrict__ b_up,              // [256]
    const unsigned short* __restrict__ wf2_b,    // [96][256] bf16
    float* __restrict__ xhat)                    // [16,128,128]
{
  __shared__ __attribute__((aligned(16))) unsigned short a2s[QN * A2S]; // 198 KB
  __shared__ __attribute__((aligned(16))) unsigned short ssl[QN * SSS]; //  78 KB

  const int tid  = threadIdx.x;
  const int lane = tid & 31;
  const int wave = tid >> 5;      // 0..7
  const int pair = wave >> 1;     // 0..3 : m-tile group
  const int sub  = wave & 1;      // n-half
  const int half = lane >> 4;
  const int l15  = lane & 15;

  const int x0 = blockIdx.x * 8;   // 16 x-tiles
  const int y0 = blockIdx.y * 16;  //  8 y-tiles
  const int b  = blockIdx.z;       // 16 batches

  // hoist b_up for this wave's columns
  float bupv[8];
#pragma unroll
  for (int nt = 0; nt < 8; nt++) bupv[nt] = b_up[sub * 128 + nt * 16 + l15];

  // ---------------- GEMM1: a2 (M=384, N=256, K=128) ----------------
  for (int i = 0; i < 6; i++) {
    int mb = pair + 4 * i;              // m-tile 0..23
    int q  = mb * 16 + l15;             // this lane's halo pixel (row)
    int qy = q >> 4, qx = q & 15;
    int gy = y0 + qy - 4, gx = x0 + qx - 4;
    int inimg = (gy >= 0) & (gy < HW) & (gx >= 0) & (gx < HW);
    int gp = (b * HW + (inimg ? gy : 0)) * HW + (inimg ? gx : 0);
    int   idxq = inimg ? idx_arr[gp] : 0;
    float vq   = inimg ? val_arr[gp] : 0.f;
    const float* wrow = wdt + idxq * DCH;

    FragC acc[8];
#pragma unroll
    for (int nt = 0; nt < 8; nt++)
#pragma unroll
      for (int r = 0; r < 8; r++) acc[nt].f[r] = 0.f;

    for (int ks = 0; ks < 4; ks++) {
      int d0 = ks * 32 + 8 * half;
      FragAB afr;                      // h values for this lane's (q, d) slots
#pragma unroll
      for (int g = 0; g < 2; g++) {
        int db = d0 + 16 * g;
#pragma unroll
        for (int jj = 0; jj < 4; jj++) {
          float h0 = fmaxf(fmaf(vq, wrow[db + 2 * jj],     b_down[db + 2 * jj]),     0.f);
          float h1 = fmaxf(fmaf(vq, wrow[db + 2 * jj + 1], b_down[db + 2 * jj + 1]), 0.f);
          afr.u[g * 4 + jj] = pack2(f2bf(h0), f2bf(h1));
        }
      }
#pragma unroll
      for (int nt = 0; nt < 8; nt++) {
        int ch = sub * 128 + nt * 16 + l15;
        const unsigned short* p = wup_b + ch * DCH + ks * 32 + 8 * half;
        FragAB bfr;
        bfr.q[0] = *(const u32x4*)p;
        bfr.q[1] = *(const u32x4*)(p + 16);
        wmma_bf16(acc[nt], afr, bfr);
      }
    }
#pragma unroll
    for (int nt = 0; nt < 8; nt++) {
      int ch = sub * 128 + nt * 16 + l15;
#pragma unroll
      for (int r = 0; r < 8; r++) {
        int qq = mb * 16 + r + 8 * half;
        a2s[qq * A2S + ch] = f2bf(fmaxf(acc[nt].f[r] + bupv[nt], 0.f));
      }
    }
  }
  __syncthreads();

  // ---------------- GEMM2: s (M=384, N=96, K=256) ----------------
  for (int i = 0; i < 6; i++) {
    int mb = pair + 4 * i;
    FragC acc[3];
#pragma unroll
    for (int nt = 0; nt < 3; nt++)
#pragma unroll
      for (int r = 0; r < 8; r++) acc[nt].f[r] = 0.f;

    for (int ks = 0; ks < 8; ks++) {
      const unsigned short* p = &a2s[(mb * 16 + l15) * A2S + ks * 32 + 8 * half];
      FragAB afr;
      afr.q[0] = *(const u32x4*)p;
      afr.q[1] = *(const u32x4*)(p + 16);
#pragma unroll
      for (int nt = 0; nt < 3; nt++) {
        int t = sub * 48 + nt * 16 + l15;
        const unsigned short* pb = wf2_b + t * KCH + ks * 32 + 8 * half;
        FragAB bfr;
        bfr.q[0] = *(const u32x4*)pb;
        bfr.q[1] = *(const u32x4*)(pb + 16);
        wmma_bf16(acc[nt], afr, bfr);
      }
    }
#pragma unroll
    for (int nt = 0; nt < 3; nt++) {
      int t = sub * 48 + nt * 16 + l15;
#pragma unroll
      for (int r = 0; r < 8; r++) {
        int qq = mb * 16 + r + 8 * half;
        ssl[qq * SSS + t] = f2bf(acc[nt].f[r]);
      }
    }
  }
  __syncthreads();

  // ---------------- gather: x_hat[p] = sum_t s[p + off_t, t] ----------------
  if (tid < 128) {
    int oy = tid >> 3, ox = tid & 7;
    float sum = 0.f;
    for (int ry = 0; ry < 9; ry++) {
      int gy = y0 + oy + ry - 4;
      if (gy < 0 || gy >= HW) continue;
      for (int rx = 0; rx < 9; rx++) {
        int gx = x0 + ox + rx - 4;
        if (gx < 0 || gx >= HW) continue;
        int qq = (oy + ry) * QW + (ox + rx);
        sum += bf2f(ssl[qq * SSS + ry * 9 + rx]);
      }
    }
    xhat[(b * HW + (y0 + oy)) * HW + (x0 + ox)] = sum;
  }
}

// ---------------------------------------------------------------------------
// Kernel 3: usage EMA
// ---------------------------------------------------------------------------
__global__ void usage_kernel(const unsigned int* __restrict__ ucnt,
                             const float* __restrict__ ema,
                             float* __restrict__ out) {
  int k = threadIdx.x;
  float p = (float)ucnt[k] / ((float)NPIX + 1e-6f);
  out[k] = ema[k] * 0.99f + p * 0.01f;
}

// ---------------------------------------------------------------------------
extern "C" void kernel_launch(void* const* d_in, const int* in_sizes, int n_in,
                              void* d_out, int out_size, void* d_ws, size_t ws_size,
                              hipStream_t stream) {
  const float* x      = (const float*)d_in[0];
  const float* w_conv = (const float*)d_in[1];
  const float* gate   = (const float*)d_in[2];
  const float* w_down = (const float*)d_in[3];
  const float* b_down = (const float*)d_in[4];
  const float* w_up   = (const float*)d_in[5];
  const float* b_up   = (const float*)d_in[6];
  const float* ema    = (const float*)d_in[7];

  char* ws = (char*)d_ws;
  size_t off = 0;
  auto alloc = [&](size_t bytes) {
    void* p = ws + off;
    off = (off + bytes + 255) & ~(size_t)255;
    return p;
  };
  unsigned short* wenc_b = (unsigned short*)alloc(KCH * TPAD * 2);
  unsigned short* wup_b  = (unsigned short*)alloc(KCH * DCH * 2);
  unsigned short* wf2_b  = (unsigned short*)alloc(TPAD * KCH * 2);
  float*          wdt    = (float*)alloc(KCH * DCH * 4);
  float*          sigg   = (float*)alloc(KCH * 4);
  unsigned int*   ucnt   = (unsigned int*)alloc(KCH * 4);
  int*            idx_a  = (int*)alloc(NPIX * 4);
  float*          val_a  = (float*)alloc(NPIX * 4);
  (void)ws_size; (void)in_sizes; (void)n_in; (void)out_size;

  float* xhat      = (float*)d_out;
  float* usage_out = (float*)d_out + NPIX;

  prep_kernel<<<64, 256, 0, stream>>>(w_conv, gate, w_down, w_up,
                                      wenc_b, wup_b, wf2_b, wdt, sigg, ucnt);
  enc_kernel<<<dim3(4, HW, BATCH), 128, 0, stream>>>(x, wenc_b, sigg,
                                                     idx_a, val_a, ucnt);
  usage_kernel<<<1, KCH, 0, stream>>>(ucnt, ema, usage_out);
  dec_kernel<<<dim3(16, 8, BATCH), 256, 0, stream>>>(idx_a, val_a, wdt, b_down,
                                                     wup_b, b_up, wf2_b, xhat);
}
